// MambaLayer_47553877901710
// MI455X (gfx1250) — matrix operand order repaired and checked by reading, not compiled, not run on hardware
//
#include <hip/hip_runtime.h>
#include <hip/hip_bf16.h>
#include <math.h>

// ---------------------------------------------------------------------------
// Mamba layer for MI455X (gfx1250, wave32, WMMA).
// GEMMs run on v_wmma_f32_16x16x32_f16 (f16 in, f32 accum).
// ---------------------------------------------------------------------------

typedef __attribute__((ext_vector_type(16))) _Float16 v16h;
typedef __attribute__((ext_vector_type(8)))  _Float16 v8h;
typedef __attribute__((ext_vector_type(8)))  float    v8f;

#define BATCH 2
#define DIMC  128
#define LSP   4096            // 16*16*16 spatial positions
#define DI    256             // D_INNER
#define DSTATE 16
#define DTR   8
#define ROWS  (BATCH * LSP)   // 8192 tokens
#define NPAD  48              // x_proj N (40) padded to 3 WMMA tiles

// ---------------------------------------------------------------------------
// small elementwise helpers
// ---------------------------------------------------------------------------
__global__ void k_f32_to_f16(const float* __restrict__ src,
                             _Float16* __restrict__ dst, int n) {
  int i = blockIdx.x * blockDim.x + threadIdx.x;
  if (i < n) dst[i] = (_Float16)src[i];
}

__global__ void k_zero_f16(_Float16* __restrict__ dst, int n) {
  int i = blockIdx.x * blockDim.x + threadIdx.x;
  if (i < n) dst[i] = (_Float16)0.0f;
}

// x: (B, DIMC, L) f32  ->  xT: (B, L, DIMC) f16
__global__ void k_transpose_x(const float* __restrict__ x,
                              _Float16* __restrict__ xT) {
  int i = blockIdx.x * blockDim.x + threadIdx.x;  // over B*DIMC*L
  if (i >= BATCH * DIMC * LSP) return;
  int pp = i & (LSP - 1);
  int c  = (i >> 12) & (DIMC - 1);
  int b  = i >> 19;
  xT[((size_t)b * LSP + pp) * DIMC + c] = (_Float16)x[i];
}

// ---------------------------------------------------------------------------
// WMMA tile core: one wave computes a 16x16 f32 tile of  C = A * W^T
//   A : M x K  f16 row-major (lda)
//   W : N x K  f16 row-major (ldw)   (weights stored out x in)
// Fragment layouts per CDNA5 ISA 7.12.2 (wave32).
// ---------------------------------------------------------------------------
template <int K>
__device__ inline v8f wmma_tile_AWt(const _Float16* __restrict__ A, int lda,
                                    const _Float16* __restrict__ W, int ldw,
                                    int m0, int n0, int lane) {
  const int hv  = lane >> 4;   // 0: lanes 0-15, 1: lanes 16-31
  const int l16 = lane & 15;
  const _Float16* arow = A + (size_t)(m0 + l16) * lda;  // row M = l16 (both halves)
  const _Float16* wrow = W + (size_t)(n0 + l16) * ldw;  // col N = l16 (both halves)
  v8f acc = {};
#pragma unroll
  for (int k0 = 0; k0 < K; k0 += 32) {
    // A 16x32: half0 -> K {k0..k0+7, k0+16..k0+23}; half1 -> +8 within each group
    v8h a0 = *(const v8h*)(arow + k0 + hv * 8);
    v8h a1 = *(const v8h*)(arow + k0 + 16 + hv * 8);
    // B 32x16 (= W^T tile): same K layout as A
    v8h b0 = *(const v8h*)(wrow + k0 + hv * 8);
    v8h b1 = *(const v8h*)(wrow + k0 + 16 + hv * 8);
    v16h a, b;
#pragma unroll
    for (int i = 0; i < 8; ++i) {
      a[i] = a0[i]; a[i + 8] = a1[i];
      b[i] = b0[i]; b[i + 8] = b1[i];
    }
    acc = __builtin_amdgcn_wmma_f32_16x16x32_f16(
        /*neg_a=*/false, a, /*neg_b=*/false, b,
        /*c_mod=*/(short)0, acc, /*reuse_a=*/false, /*reuse_b=*/false);
  }
  return acc;
}

// Generic GEMM: 256 threads = 8 waves; wave w -> M-tile (blockIdx.x*8+w),
// blockIdx.y -> N-tile. C row-major f32 (ldc).
template <int K>
__global__ void k_gemm_f16(const _Float16* __restrict__ A, int lda,
                           const _Float16* __restrict__ W, int ldw,
                           float* __restrict__ C, int ldc) {
  const int wave = threadIdx.x >> 5;
  const int lane = threadIdx.x & 31;
  const int m0 = (blockIdx.x * 8 + wave) * 16;
  const int n0 = blockIdx.y * 16;
  v8f acc = wmma_tile_AWt<K>(A, lda, W, ldw, m0, n0, lane);
  const int hv = lane >> 4, l16 = lane & 15;
  float* crow = C + (size_t)(m0 + hv * 8) * ldc + n0 + l16;
#pragma unroll
  for (int v = 0; v < 8; ++v) crow[(size_t)v * ldc] = acc[v];
}

// out_proj GEMM with fused residual + layout epilogue:
//   out[b, n, p] = (yact @ out_proj_w^T)[b*L+p, n] + xn_res[b, n, p]
__global__ void k_gemm_out(const _Float16* __restrict__ A,        // yact (ROWS x DI)
                           const _Float16* __restrict__ W,        // out_proj (DIMC x DI)
                           const float* __restrict__ xn_res,      // (B, DIMC, L)
                           float* __restrict__ out) {             // (B, DIMC, L)
  const int wave = threadIdx.x >> 5;
  const int lane = threadIdx.x & 31;
  const int m0 = (blockIdx.x * 8 + wave) * 16;
  const int n0 = blockIdx.y * 16;
  v8f acc = wmma_tile_AWt<DI>(A, DI, W, DI, m0, n0, lane);
  const int hv = lane >> 4, l16 = lane & 15;
  const int ng = n0 + l16;
#pragma unroll
  for (int v = 0; v < 8; ++v) {
    int m = m0 + hv * 8 + v;           // global token row
    int b = m >> 12;                   // / LSP
    int pp = m & (LSP - 1);
    size_t oidx = ((size_t)b * DIMC + ng) * LSP + pp;
    out[oidx] = acc[v] + xn_res[oidx];
  }
}

// ---------------------------------------------------------------------------
// BatchNorm: stats then apply+relu
// ---------------------------------------------------------------------------
__global__ void k_bn_stats(const float* __restrict__ xnraw,  // (ROWS, DIMC)
                           float* __restrict__ stats) {      // [0..127]=mean, [128..255]=rstd
  const int o = blockIdx.x;
  __shared__ float ss[256], ss2[256];
  float s = 0.f, s2 = 0.f;
  for (int r = threadIdx.x; r < ROWS; r += blockDim.x) {
    float v = xnraw[(size_t)r * DIMC + o];
    s += v; s2 += v * v;
  }
  ss[threadIdx.x] = s; ss2[threadIdx.x] = s2;
  __syncthreads();
  for (int off = 128; off > 0; off >>= 1) {
    if (threadIdx.x < off) {
      ss[threadIdx.x]  += ss[threadIdx.x + off];
      ss2[threadIdx.x] += ss2[threadIdx.x + off];
    }
    __syncthreads();
  }
  if (threadIdx.x == 0) {
    float mean = ss[0] * (1.0f / ROWS);
    float var  = ss2[0] * (1.0f / ROWS) - mean * mean;
    stats[o] = mean;
    stats[DIMC + o] = rsqrtf(var + 1e-5f);
  }
}

__global__ void k_bn_apply(const float* __restrict__ xnraw,   // (ROWS, DIMC)
                           const float* __restrict__ stats,
                           const float* __restrict__ gamma,
                           const float* __restrict__ beta,
                           float* __restrict__ xn_res,        // (B, DIMC, L)
                           _Float16* __restrict__ xf_h) {     // (ROWS, DIMC)
  int i = blockIdx.x * blockDim.x + threadIdx.x;
  if (i >= ROWS * DIMC) return;
  int o = i & (DIMC - 1);
  int row = i >> 7;
  float t = (xnraw[i] - stats[o]) * stats[DIMC + o] * gamma[o] + beta[o];
  t = fmaxf(t, 0.0f);
  xf_h[i] = (_Float16)t;
  int b = row >> 12, pp = row & (LSP - 1);
  xn_res[((size_t)b * DIMC + o) * LSP + pp] = t;
}

// ---------------------------------------------------------------------------
// depthwise causal conv (D_CONV=4) + SiLU; xi = xz[:, :DI]
// ---------------------------------------------------------------------------
__global__ void k_conv_silu(const float* __restrict__ xz,    // (ROWS, 2*DI)
                            const float* __restrict__ conv_w,// (DI, 4)
                            const float* __restrict__ conv_b,
                            float* __restrict__ xc_f,        // (ROWS, DI)
                            _Float16* __restrict__ xc_h) {
  int i = blockIdx.x * blockDim.x + threadIdx.x;
  if (i >= ROWS * DI) return;
  int d = i & (DI - 1);
  int row = i >> 8;
  int b = row >> 12, l = row & (LSP - 1);
  float acc = conv_b[d];
#pragma unroll
  for (int k = 0; k < 4; ++k) {
    int l2 = l + k - 3;
    if (l2 >= 0)
      acc += conv_w[d * 4 + k] * xz[((size_t)(b * LSP + l2)) * (2 * DI) + d];
  }
  float s = acc / (1.0f + expf(-acc));   // SiLU
  xc_f[i] = s;
  xc_h[i] = (_Float16)s;
}

// ---------------------------------------------------------------------------
// delta = softplus(dt_r @ dt_proj_w^T + dt_proj_b), dt_r = dbl[:, :8]
// ---------------------------------------------------------------------------
__global__ void k_delta(const float* __restrict__ dbl,       // (ROWS, NPAD)
                        const float* __restrict__ dt_proj_w, // (DI, DTR)
                        const float* __restrict__ dt_proj_b,
                        float* __restrict__ delta) {         // (ROWS, DI)
  int i = blockIdx.x * blockDim.x + threadIdx.x;
  if (i >= ROWS * DI) return;
  int d = i & (DI - 1);
  int row = i >> 8;
  float acc = dt_proj_b[d];
#pragma unroll
  for (int r = 0; r < DTR; ++r)
    acc += dbl[(size_t)row * NPAD + r] * dt_proj_w[d * DTR + r];
  delta[i] = (acc > 20.0f) ? acc : log1pf(expf(acc));
}

// ---------------------------------------------------------------------------
// Selective scan. 32 blocks x 256 threads; wave32: lanes 0-15 = states of
// channel 2w, lanes 16-31 = channel 2w+1. h.C reduced by 16-lane butterfly.
// ---------------------------------------------------------------------------
__global__ void k_scan(const float* __restrict__ delta,   // (ROWS, DI)
                       const float* __restrict__ dbl,     // (ROWS, NPAD): B@8, C@24
                       const float* __restrict__ xc_f,    // (ROWS, DI)
                       const float* __restrict__ xz,      // (ROWS, 2*DI): z @ DI
                       const float* __restrict__ A_log,   // (DI, DSTATE)
                       const float* __restrict__ D_skip,  // (DI)
                       _Float16* __restrict__ yact) {     // (ROWS, DI)
  const int tid = threadIdx.x;
  const int n = tid & 15;
  const int b = blockIdx.x >> 4;
  const int d = ((blockIdx.x & 15) << 4) + (tid >> 4);
  const float And = -expf(A_log[d * DSTATE + n]);
  const float Dsk = D_skip[d];
  const size_t rowbase = (size_t)b * LSP;
  float h = 0.0f;
  for (int l = 0; l < LSP; ++l) {
    const size_t row = rowbase + l;
    float dl  = delta[row * DI + d];
    float Bn  = dbl[row * NPAD + DTR + n];
    float Cn  = dbl[row * NPAD + DTR + DSTATE + n];
    float xcv = xc_f[row * DI + d];
    h = h * expf(dl * And) + dl * Bn * xcv;
    float part = h * Cn;
    part += __shfl_xor(part, 1, 32);
    part += __shfl_xor(part, 2, 32);
    part += __shfl_xor(part, 4, 32);
    part += __shfl_xor(part, 8, 32);
    if (n == 0) {
      float zv = xz[row * (2 * DI) + DI + d];
      float y = (part + xcv * Dsk) * (zv / (1.0f + expf(-zv)));
      yact[row * DI + d] = (_Float16)y;
    }
  }
}

// ---------------------------------------------------------------------------
// host launcher
// ---------------------------------------------------------------------------
static inline int cdiv(int a, int b) { return (a + b - 1) / b; }

extern "C" void kernel_launch(void* const* d_in, const int* in_sizes, int n_in,
                              void* d_out, int out_size, void* d_ws, size_t ws_size,
                              hipStream_t stream) {
  const float* x          = (const float*)d_in[0];
  const float* nin_w      = (const float*)d_in[1];
  const float* bn_gamma   = (const float*)d_in[2];
  const float* bn_beta    = (const float*)d_in[3];
  const float* in_proj_w  = (const float*)d_in[4];
  const float* conv_w     = (const float*)d_in[5];
  const float* conv_b     = (const float*)d_in[6];
  const float* x_proj_w   = (const float*)d_in[7];
  const float* dt_proj_w  = (const float*)d_in[8];
  const float* dt_proj_b  = (const float*)d_in[9];
  const float* A_log      = (const float*)d_in[10];
  const float* D_skip     = (const float*)d_in[11];
  const float* out_proj_w = (const float*)d_in[12];
  float* out = (float*)d_out;

  // workspace carve-out (256B aligned regions)
  char* wp = (char*)d_ws;
  auto carve = [&](size_t bytes) -> void* {
    void* r = (void*)wp;
    wp += (bytes + 255) & ~(size_t)255;
    return r;
  };
  _Float16* w_nin_h  = (_Float16*)carve((size_t)DIMC * DIMC * 2);
  _Float16* w_inp_h  = (_Float16*)carve((size_t)(2 * DI) * DIMC * 2);
  _Float16* w_xp_h   = (_Float16*)carve((size_t)NPAD * DI * 2);
  _Float16* w_outp_h = (_Float16*)carve((size_t)DIMC * DI * 2);
  _Float16* xT_h     = (_Float16*)carve((size_t)ROWS * DIMC * 2);
  _Float16* xf_h     = (_Float16*)carve((size_t)ROWS * DIMC * 2);
  _Float16* xc_h     = (_Float16*)carve((size_t)ROWS * DI * 2);
  _Float16* yact_h   = (_Float16*)carve((size_t)ROWS * DI * 2);
  float* xnraw   = (float*)carve((size_t)ROWS * DIMC * 4);
  float* xn_res  = (float*)carve((size_t)ROWS * DIMC * 4);
  float* stats   = (float*)carve((size_t)2 * DIMC * 4);
  float* xz      = (float*)carve((size_t)ROWS * 2 * DI * 4);
  float* xc_f    = (float*)carve((size_t)ROWS * DI * 4);
  float* dbl     = (float*)carve((size_t)ROWS * NPAD * 4);
  float* delta   = (float*)carve((size_t)ROWS * DI * 4);

  const int T = 256;

  // 1) weight conversions (x_proj padded to 48 rows, zero-filled)
  k_f32_to_f16<<<cdiv(DIMC * DIMC, T), T, 0, stream>>>(nin_w, w_nin_h, DIMC * DIMC);
  k_f32_to_f16<<<cdiv(2 * DI * DIMC, T), T, 0, stream>>>(in_proj_w, w_inp_h, 2 * DI * DIMC);
  k_zero_f16<<<cdiv(NPAD * DI, T), T, 0, stream>>>(w_xp_h, NPAD * DI);
  k_f32_to_f16<<<cdiv((DTR + 2 * DSTATE) * DI, T), T, 0, stream>>>(x_proj_w, w_xp_h,
                                                                   (DTR + 2 * DSTATE) * DI);
  k_f32_to_f16<<<cdiv(DIMC * DI, T), T, 0, stream>>>(out_proj_w, w_outp_h, DIMC * DI);

  // 2) transpose x -> token-major f16
  k_transpose_x<<<cdiv(BATCH * DIMC * LSP, T), T, 0, stream>>>(x, xT_h);

  // 3) nin GEMM: xnraw[ROWS, 128] = xT @ nin_w^T       (WMMA)
  k_gemm_f16<DIMC><<<dim3(ROWS / 16 / 8, DIMC / 16), T, 0, stream>>>(
      xT_h, DIMC, w_nin_h, DIMC, xnraw, DIMC);

  // 4) BN stats + apply (relu); writes residual (B,C,L) and xf f16
  k_bn_stats<<<DIMC, T, 0, stream>>>(xnraw, stats);
  k_bn_apply<<<cdiv(ROWS * DIMC, T), T, 0, stream>>>(xnraw, stats, bn_gamma, bn_beta,
                                                     xn_res, xf_h);

  // 5) in_proj GEMM: xz[ROWS, 512] = xf @ in_proj_w^T  (WMMA)
  k_gemm_f16<DIMC><<<dim3(ROWS / 16 / 8, (2 * DI) / 16), T, 0, stream>>>(
      xf_h, DIMC, w_inp_h, DIMC, xz, 2 * DI);

  // 6) depthwise causal conv + SiLU
  k_conv_silu<<<cdiv(ROWS * DI, T), T, 0, stream>>>(xz, conv_w, conv_b, xc_f, xc_h);

  // 7) x_proj GEMM: dbl[ROWS, 48] = xc @ x_proj_w^T    (WMMA, padded N)
  k_gemm_f16<DI><<<dim3(ROWS / 16 / 8, NPAD / 16), T, 0, stream>>>(
      xc_h, DI, w_xp_h, DI, dbl, NPAD);

  // 8) delta = softplus(dt_r @ dt_proj_w^T + b)
  k_delta<<<cdiv(ROWS * DI, T), T, 0, stream>>>(dbl, dt_proj_w, dt_proj_b, delta);

  // 9) selective scan -> y * silu(z) as f16 GEMM operand
  k_scan<<<BATCH * (DI / 16), T, 0, stream>>>(delta, dbl, xc_f, xz, A_log, D_skip, yact_h);

  // 10) out_proj GEMM + residual, final layout            (WMMA)
  k_gemm_out<<<dim3(ROWS / 16 / 8, DIMC / 16), T, 0, stream>>>(yact_h, w_outp_h, xn_res, out);
}
